// TrajAR_decoder_Attention_88416196756299
// MI455X (gfx1250) — compile-verified, matrix operand order checked by reading, not run on hardware
//
#include <hip/hip_runtime.h>
#include <cstdint>
#include <cstddef>

#define B_  4
#define S_  2048
#define D_  1024
#define H_  16
#define HD_ 64

typedef __attribute__((ext_vector_type(16))) __bf16 v16bf;
typedef __attribute__((ext_vector_type(8)))  float  v8f;

// Load a 16-element bf16 fragment as two contiguous 16-byte chunks.
static __device__ __forceinline__ v16bf load_frag(const __bf16* p0, const __bf16* p1) {
    union { v16bf v; float4 f[2]; } u;
    u.f[0] = *reinterpret_cast<const float4*>(p0);
    u.f[1] = *reinterpret_cast<const float4*>(p1);
    return u.v;
}

static __device__ __forceinline__ v8f wmma_bf16(v16bf a, v16bf b, v8f c) {
    // D = A(16x32 bf16) * B(32x16 bf16) + C(16x16 f32)
    return __builtin_amdgcn_wmma_f32_16x16x32_bf16(
        /*neg_a=*/false, a, /*neg_b=*/false, b,
        /*c_mod=*/(short)0, c, /*reuse_a=*/false, /*reuse_b=*/false);
}

// Async global -> LDS copy of 16 bytes per lane (CDNA5, tracked by ASYNCcnt).
static __device__ __forceinline__ void async_ld16(__bf16* lds_dst, const __bf16* gsrc) {
    unsigned int l = (unsigned int)(size_t)lds_dst;   // low 32 bits = LDS byte offset
    asm volatile("global_load_async_to_lds_b128 %0, %1, off"
                 :: "v"(l), "v"(gsrc) : "memory");
}
static __device__ __forceinline__ void wait_async0() {
    asm volatile("s_wait_asynccnt 0x0" ::: "memory");
}

// Reductions across the 16 lanes that share a C-matrix row (wave32, groups of 16).
static __device__ __forceinline__ float red_max16(float x) {
#pragma unroll
    for (int i = 8; i >= 1; i >>= 1) x = fmaxf(x, __shfl_xor(x, i, 16));
    return x;
}
static __device__ __forceinline__ float red_sum16(float x) {
#pragma unroll
    for (int i = 8; i >= 1; i >>= 1) x += __shfl_xor(x, i, 16);
    return x;
}

// ---------------------------------------------------------------- convert x
__global__ __launch_bounds__(256) void cvt_bf16_kernel(const float* __restrict__ in,
                                                       __bf16* __restrict__ out, int n) {
    int i = blockIdx.x * 256 + threadIdx.x;
    if (i < n) out[i] = (__bf16)in[i];
}

// ------------------------------------------------- pack mask int32 -> bits
// pm[w] bit j = (mask[w*32+j] != 0). One word per wave via ballot.
__global__ __launch_bounds__(256) void pack_mask_kernel(const int* __restrict__ mask,
                                                        unsigned int* __restrict__ pm,
                                                        int nwords) {
    int w = blockIdx.x * 8 + (threadIdx.x >> 5);
    int lane = threadIdx.x & 31;
    if (w < nwords) {
        int v = mask[(size_t)w * 32 + lane];
        unsigned long long b = __ballot(v != 0);
        if (lane == 0) pm[w] = (unsigned int)b;
    }
}

// ------------------------------------------- transpose+convert weights W->Wt
// Wt[n][k] = (bf16) W[k][n], D_ x D_.  Block (32,8), grid (32,32,4).
__global__ __launch_bounds__(256) void transpose_w_kernel(
    const float* __restrict__ w0, const float* __restrict__ w1,
    const float* __restrict__ w2, const float* __restrict__ w3,
    __bf16* __restrict__ t0, __bf16* __restrict__ t1,
    __bf16* __restrict__ t2, __bf16* __restrict__ t3) {
    __shared__ float tile[32][33];
    const float* W = (blockIdx.z == 0) ? w0 : (blockIdx.z == 1) ? w1
                   : (blockIdx.z == 2) ? w2 : w3;
    __bf16* T      = (blockIdx.z == 0) ? t0 : (blockIdx.z == 1) ? t1
                   : (blockIdx.z == 2) ? t2 : t3;
    const int tx = threadIdx.x, ty = threadIdx.y;
    const int n0 = blockIdx.x * 32, k0 = blockIdx.y * 32;
#pragma unroll
    for (int j = 0; j < 4; ++j)
        tile[ty + 8 * j][tx] = W[(size_t)(k0 + ty + 8 * j) * D_ + (n0 + tx)];
    __syncthreads();
#pragma unroll
    for (int j = 0; j < 4; ++j)
        T[(size_t)(n0 + ty + 8 * j) * D_ + (k0 + tx)] = (__bf16)tile[tx][ty + 8 * j];
}

// ------------------------------------------------------------- WMMA GEMM
// C[m][n] = sum_k A[m][k]*Bt[n][k] + bias[n].
// MODE 0: store bf16 head-split [B,H,S,HD]   (Q,K)
// MODE 1: store f32 row-major   [M,D]        (final output)
// MODE 2: store bf16 transposed [B,H,HD,S]   (V)
template <int MODE>
__global__ __launch_bounds__(256) void gemm_kernel(
    const __bf16* __restrict__ A, const __bf16* __restrict__ Bt,
    const float* __restrict__ bias, void* __restrict__ outv, int K) {
    const int wave = threadIdx.x >> 5, lane = threadIdx.x & 31;
    const int row = lane & 15, half = lane >> 4, kb = half * 8;
    const int m0 = blockIdx.x * 32 + (wave & 1) * 16;
    const int n0 = blockIdx.y * 256 + (wave >> 1) * 64;

    v8f c0 = {}, c1 = {}, c2 = {}, c3 = {};
    const __bf16* ap  = A  + (size_t)(m0 + row) * K;
    const __bf16* bp0 = Bt + (size_t)(n0 + row) * K;
    const __bf16* bp1 = bp0 + (size_t)16 * K;
    const __bf16* bp2 = bp0 + (size_t)32 * K;
    const __bf16* bp3 = bp0 + (size_t)48 * K;

    for (int k = 0; k < K; k += 32) {
        // Load all fragments first so the backend can clause the b128 loads
        // and overlap them with the WMMA pipe.
        v16bf af  = load_frag(ap  + k + kb, ap  + k + kb + 16);
        v16bf bf0 = load_frag(bp0 + k + kb, bp0 + k + kb + 16);
        v16bf bf1 = load_frag(bp1 + k + kb, bp1 + k + kb + 16);
        v16bf bf2 = load_frag(bp2 + k + kb, bp2 + k + kb + 16);
        v16bf bf3 = load_frag(bp3 + k + kb, bp3 + k + kb + 16);
        c0 = wmma_bf16(af, bf0, c0);
        c1 = wmma_bf16(af, bf1, c1);
        c2 = wmma_bf16(af, bf2, c2);
        c3 = wmma_bf16(af, bf3, c3);
    }

    const float bb[4] = { bias[n0 + row], bias[n0 + 16 + row],
                          bias[n0 + 32 + row], bias[n0 + 48 + row] };
    v8f cc[4] = { c0, c1, c2, c3 };
#pragma unroll
    for (int nt = 0; nt < 4; ++nt) {
#pragma unroll
        for (int r = 0; r < 8; ++r) {
            float val = cc[nt][r] + bb[nt];
            int m = m0 + r + half * 8;           // global row
            int n = n0 + nt * 16 + row;          // global col
            if (MODE == 1) {
                ((float*)outv)[(size_t)m * D_ + n] = val;
            } else {
                int b = m >> 11, s = m & (S_ - 1);
                int h = n >> 6,  hd = n & (HD_ - 1);
                size_t idx = (MODE == 0)
                    ? ((size_t)((b * H_ + h) * S_ + s) * HD_ + hd)
                    : ((size_t)((b * H_ + h) * HD_ + hd) * S_ + s);
                ((__bf16*)outv)[idx] = (__bf16)val;
            }
        }
    }
}

// ------------------------------------------------------- flash attention
// Block = 8 waves, all on the same (b,h); wave w owns q-tile (qb*8 + w).
// Per 32-key step the whole block cooperatively async-copies the K tile
// (32x64) and V^T tile (64x32) into double-buffered LDS, then each wave:
//   S = Q(16x64) @ K^T  (4 wmma)  ->  online softmax (packed-bit mask)
//   O = O*alpha + P(16x32) @ V    (4 wmma)
__global__ __launch_bounds__(256) void attn_kernel(
    const __bf16* __restrict__ q, const __bf16* __restrict__ k,
    const __bf16* __restrict__ vt, const unsigned int* __restrict__ pm,
    __bf16* __restrict__ ctx) {
    __shared__ __align__(16) __bf16 kbuf[2][32 * HD_];   // [key][hd]
    __shared__ __align__(16) __bf16 vbuf[2][HD_ * 32];   // [hd][key]
    __shared__ __align__(16) __bf16 plds[8][16 * 32];    // per-wave P tile

    const int wave = threadIdx.x >> 5, lane = threadIdx.x & 31;
    const int row = lane & 15, half = lane >> 4, kb = half * 8;
    const int bh = blockIdx.x >> 4;               // 0..63
    const int qb = blockIdx.x & 15;
    const int h = bh & (H_ - 1), b = bh >> 4;
    const int qt = qb * 8 + wave;                 // q-tile 0..127

    const __bf16* qp = q  + ((size_t)(b * H_ + h) * S_ + qt * 16) * HD_;
    const __bf16* kp = k  + (size_t)(b * H_ + h) * S_ * HD_;
    const __bf16* vp = vt + (size_t)(b * H_ + h) * HD_ * S_;
    const unsigned int* pmp = pm + ((size_t)b * S_ + qt * 16) * (S_ / 32);
    __bf16* pl = plds[wave];

    // Cooperative stage indices (one 16B async chunk per thread per tile).
    const int t = threadIdx.x;
    const int k_key = t >> 3, k_ch = (t & 7) * 8;   // K tile: 32 keys x 128B
    const int v_hd  = t >> 2, v_ch = (t & 3) * 8;   // V tile: 64 hd  x  64B

    const __bf16* qr = qp + row * HD_;
    const v16bf qf0 = load_frag(qr + kb,      qr + kb + 16);       // hd 0..31
    const v16bf qf1 = load_frag(qr + 32 + kb, qr + 32 + kb + 16);  // hd 32..63

    v8f o0 = {}, o1 = {}, o2 = {}, o3 = {};
    float m8[8], l8[8];
#pragma unroll
    for (int r = 0; r < 8; ++r) { m8[r] = -__builtin_inff(); l8[r] = 0.f; }

    // Prime buffer 0.
    async_ld16(&kbuf[0][k_key * HD_ + k_ch], kp + (size_t)k_key * HD_ + k_ch);
    async_ld16(&vbuf[0][v_hd * 32 + v_ch],   vp + (size_t)v_hd * S_ + v_ch);
    wait_async0();
    __syncthreads();

    for (int kt = 0; kt < S_; kt += 32) {
        const int cur = (kt >> 5) & 1;
        const bool more = (kt + 32) < S_;
        if (more) {   // prefetch next tile into the other buffer
            const int kn = kt + 32;
            async_ld16(&kbuf[cur ^ 1][k_key * HD_ + k_ch],
                       kp + (size_t)(kn + k_key) * HD_ + k_ch);
            async_ld16(&vbuf[cur ^ 1][v_hd * 32 + v_ch],
                       vp + (size_t)v_hd * S_ + kn + v_ch);
        }

        // Mask words early: latency hidden under the score WMMAs.
        unsigned int mw[8];
#pragma unroll
        for (int r = 0; r < 8; ++r)
            mw[r] = pmp[(size_t)(r + half * 8) * (S_ / 32) + (kt >> 5)];

        // ---- scores: Q @ K^T from LDS (batch loads, then WMMA chain) ----
        const __bf16* kr0 = &kbuf[cur][(size_t)row * HD_];
        const __bf16* kr1 = kr0 + (size_t)16 * HD_;
        v16bf kf00 = load_frag(kr0 + kb,      kr0 + kb + 16);
        v16bf kf01 = load_frag(kr0 + 32 + kb, kr0 + 32 + kb + 16);
        v16bf kf10 = load_frag(kr1 + kb,      kr1 + kb + 16);
        v16bf kf11 = load_frag(kr1 + 32 + kb, kr1 + 32 + kb + 16);
        v8f s0 = {}, s1 = {};
        s0 = wmma_bf16(qf0, kf00, s0);
        s0 = wmma_bf16(qf1, kf01, s0);
        s1 = wmma_bf16(qf0, kf10, s1);
        s1 = wmma_bf16(qf1, kf11, s1);

        // ---- online softmax (packed-bit mask) ----
        float fac[8];
#pragma unroll
        for (int r = 0; r < 8; ++r) {
            const int qrow = r + half * 8;
            float a = s0[r] * 0.125f;               // 1/sqrt(HD)
            float c = s1[r] * 0.125f;
            if (!((mw[r] >> row) & 1u))        a = -__builtin_inff();
            if (!((mw[r] >> (16 + row)) & 1u)) c = -__builtin_inff();
            float loc = red_max16(fmaxf(a, c));
            float nm  = fmaxf(m8[r], loc);
            float f  = __expf(m8[r] - nm); f  = (f  == f ) ? f  : 0.f;
            float pa = __expf(a - nm);     pa = (pa == pa) ? pa : 0.f;
            float pc = __expf(c - nm);     pc = (pc == pc) ? pc : 0.f;
            l8[r] = l8[r] * f + red_sum16(pa + pc);
            m8[r] = nm;
            fac[r] = f;
            pl[qrow * 32 + row]      = (__bf16)pa;  // P tile, row-major 16x32
            pl[qrow * 32 + 16 + row] = (__bf16)pc;
        }
#pragma unroll
        for (int r = 0; r < 8; ++r) {
            o0[r] *= fac[r]; o1[r] *= fac[r]; o2[r] *= fac[r]; o3[r] *= fac[r];
        }
        // LDS is in-order within a wave; block compiler reordering of the
        // bf16 stores vs. the b128 fragment reload.
        __asm__ __volatile__("" ::: "memory");
        v16bf pf = load_frag(pl + row * 32 + kb, pl + row * 32 + kb + 16);

        // ---- O += P @ V from LDS (batch loads, then WMMA chain) ----
        const __bf16* v0 = &vbuf[cur][(size_t)(row) * 32];
        const __bf16* v1 = &vbuf[cur][(size_t)(16 + row) * 32];
        const __bf16* v2 = &vbuf[cur][(size_t)(32 + row) * 32];
        const __bf16* v3 = &vbuf[cur][(size_t)(48 + row) * 32];
        v16bf vf0 = load_frag(v0 + kb, v0 + kb + 16);
        v16bf vf1 = load_frag(v1 + kb, v1 + kb + 16);
        v16bf vf2 = load_frag(v2 + kb, v2 + kb + 16);
        v16bf vf3 = load_frag(v3 + kb, v3 + kb + 16);
        o0 = wmma_bf16(pf, vf0, o0);
        o1 = wmma_bf16(pf, vf1, o1);
        o2 = wmma_bf16(pf, vf2, o2);
        o3 = wmma_bf16(pf, vf3, o3);

        if (more) wait_async0();   // next buffer filled (our own contributions)
        __syncthreads();           // all waves' contributions visible
    }

#pragma unroll
    for (int r = 0; r < 8; ++r) {
        float inv = (l8[r] > 0.f) ? (1.0f / l8[r]) : 0.f;   // fully-masked -> 0
        int srow = qt * 16 + r + half * 8;
        size_t base = ((size_t)b * S_ + srow) * D_ + h * HD_ + row;
        ctx[base]      = (__bf16)(o0[r] * inv);
        ctx[base + 16] = (__bf16)(o1[r] * inv);
        ctx[base + 32] = (__bf16)(o2[r] * inv);
        ctx[base + 48] = (__bf16)(o3[r] * inv);
    }
}

// ------------------------------------------------------------------ driver
extern "C" void kernel_launch(void* const* d_in, const int* in_sizes, int n_in,
                              void* d_out, int out_size, void* d_ws, size_t ws_size,
                              hipStream_t stream) {
    (void)in_sizes; (void)n_in; (void)out_size; (void)ws_size;
    const float* x    = (const float*)d_in[0];
    const int*   mask = (const int*)d_in[1];
    const float* Wq   = (const float*)d_in[2];
    const float* bq   = (const float*)d_in[3];
    const float* Wk   = (const float*)d_in[4];
    const float* bk   = (const float*)d_in[5];
    const float* Wv   = (const float*)d_in[6];
    const float* bv   = (const float*)d_in[7];
    const float* Wo   = (const float*)d_in[8];
    const float* bo   = (const float*)d_in[9];
    float* out = (float*)d_out;

    char* w = (char*)d_ws;
    const size_t act_bytes = (size_t)B_ * S_ * D_ * sizeof(__bf16);  // 16 MB
    const size_t w_bytes   = (size_t)D_ * D_ * sizeof(__bf16);       //  2 MB
    __bf16* xb  = (__bf16*)w; w += act_bytes;
    __bf16* wqT = (__bf16*)w; w += w_bytes;
    __bf16* wkT = (__bf16*)w; w += w_bytes;
    __bf16* wvT = (__bf16*)w; w += w_bytes;
    __bf16* woT = (__bf16*)w; w += w_bytes;
    __bf16* qh  = (__bf16*)w; w += act_bytes;   // [B,H,S,HD]
    __bf16* kh  = (__bf16*)w; w += act_bytes;   // [B,H,S,HD]
    __bf16* vvt = (__bf16*)w; w += act_bytes;   // [B,H,HD,S]
    __bf16* ctx = (__bf16*)w; w += act_bytes;   // [B,S,D]
    unsigned int* pmask = (unsigned int*)w; w += (size_t)B_ * S_ * (S_ / 32) * 4; // 2 MB

    const int n_x = B_ * S_ * D_;
    cvt_bf16_kernel<<<n_x / 256, 256, 0, stream>>>(x, xb, n_x);

    const int nwords = B_ * S_ * (S_ / 32);
    pack_mask_kernel<<<(nwords + 7) / 8, 256, 0, stream>>>(mask, pmask, nwords);

    transpose_w_kernel<<<dim3(D_ / 32, D_ / 32, 4), dim3(32, 8), 0, stream>>>(
        Wq, Wk, Wv, Wo, wqT, wkT, wvT, woT);

    const dim3 ggrid(B_ * S_ / 32, D_ / 256);
    gemm_kernel<0><<<ggrid, 256, 0, stream>>>(xb, wqT, bq, (void*)qh,  D_);
    gemm_kernel<0><<<ggrid, 256, 0, stream>>>(xb, wkT, bk, (void*)kh,  D_);
    gemm_kernel<2><<<ggrid, 256, 0, stream>>>(xb, wvT, bv, (void*)vvt, D_);

    attn_kernel<<<(B_ * H_) * 16, 256, 0, stream>>>(qh, kh, vvt, pmask, ctx);

    gemm_kernel<1><<<ggrid, 256, 0, stream>>>(ctx, woT, bo, (void*)out, D_);
}
